// PTGRUSeq2Seq_91276644974841
// MI455X (gfx1250) — matrix operand order repaired and checked by reading, not compiled, not run on hardware
//
#include <hip/hip_runtime.h>

// ---------------------------------------------------------------------------
// PTGRUSeq2Seq on gfx1250 (MI455X): persistent per-batch-tile GRU scan with
// v_wmma_f32_16x16x32_bf16 matrix cores, bf16 weights packed in B-fragment
// order, fp32 hidden state in registers + bf16 mirror in LDS for A fragments.
// Inner loops are kept rolled (#pragma unroll 1) so weight-fragment loads are
// re-issued from L2 every step instead of being hoisted/spilled to scratch.
// ---------------------------------------------------------------------------

typedef __attribute__((ext_vector_type(16))) __bf16 v16bf;
typedef __attribute__((ext_vector_type(8)))  float  v8f;

#define Hd 128
#define Gd 384
#define NBATCH 4096
#define TENC 64
#define TDEC 99
#define PKM 49152            // packed halves per 384x128 matrix (24*4*512)

// output layout (floats)
#define DEC_OUT_OFF 0
#define DEC_HID_OFF 1638400
#define ENC_OUT_OFF 3735552
#define ENC_HID_OFF 37289984
#define LOSS_OFF    39387136

__device__ __forceinline__ unsigned short f2bf(float f) {
  union { float f; unsigned u; } x; x.f = f;
  unsigned r = x.u + 0x7FFFu + ((x.u >> 16) & 1u);   // RNE
  return (unsigned short)(r >> 16);
}
__device__ __forceinline__ float sigm(float x) { return 1.0f / (1.0f + __expf(-x)); }
__device__ __forceinline__ float tanh_fast(float x) {
  float e = __expf(2.0f * x);
  return (e - 1.0f) / (e + 1.0f);
}

union Frag16 { uint4 q[2]; v16bf v; };

// B fragment: packed block (tile,kc) = 32 lanes x 16 halves, lane-major.
__device__ __forceinline__ v16bf ldB(const unsigned short* pk, int tile, int kc, int lane) {
  Frag16 u;
  const uint4* p = (const uint4*)(pk + ((tile * 4 + kc) << 9) + (lane << 4));
  u.q[0] = p[0];
  u.q[1] = p[1];
  return u.v;
}
// A fragment from bf16 LDS row (row-major [M][128]); kb = kc*32 + (lane>=16?16:0)
__device__ __forceinline__ v16bf ldA(const unsigned short* row, int kb) {
  Frag16 u;
  u.q[0] = *(const uint4*)(row + kb);
  u.q[1] = *(const uint4*)(row + kb + 8);
  return u.v;
}
#define WMMA_BF16(a, b, c) \
  __builtin_amdgcn_wmma_f32_16x16x32_bf16(false, (a), false, (b), (short)0, (c), false, false)

// ---------------------------------------------------------------------------
// Pack a row-major (N x 128) fp32 matrix W (gate = W[n][k], i.e. B = W^T)
// into bf16 B-fragment blocks: block(ntile,kc) -> lane -> 16 halves.
// lane = (n&15) | (((k>>4)&1)<<4), elem = k&15.
// ---------------------------------------------------------------------------
__global__ void pack_w_kernel(const float* __restrict__ src,
                              unsigned short* __restrict__ dst, int total) {
  int gid = blockIdx.x * blockDim.x + threadIdx.x;
  if (gid >= total) return;
  int n = gid >> 7;
  int k = gid & 127;
  int ntile = n >> 4;
  int kc = k >> 5;
  int lane = (n & 15) | (((k >> 4) & 1) << 4);
  int elem = k & 15;
  dst[((ntile * 4 + kc) << 9) + (lane << 4) + elem] = f2bf(src[gid]);
}

__global__ void zero_loss_kernel(float* p) {
  if (threadIdx.x == 0 && blockIdx.x == 0) *p = 0.0f;
}

// ---------------------------------------------------------------------------
// Main persistent kernel: one workgroup per 16 batch rows.
// ---------------------------------------------------------------------------
__global__ void __launch_bounds__(256)
gru_seq2seq_kernel(const float* __restrict__ inp,       // (4096,64,4)
                   const float* __restrict__ tgt,       // (4096,100,4)
                   const float* __restrict__ enc_wih0,  // (384,4)
                   const float* __restrict__ enc_bih,   // (4,384)
                   const float* __restrict__ enc_bhh,   // (4,384)
                   const float* __restrict__ dec_wih0,  // (384,4)
                   const float* __restrict__ dec_bih,   // (4,384)
                   const float* __restrict__ dec_bhh,   // (4,384)
                   const float* __restrict__ lin1_b,    // (128)
                   const float* __restrict__ lin2_w,    // (4,128)
                   const float* __restrict__ lin2_b,    // (4)
                   const unsigned short* __restrict__ wpk,
                   float* __restrict__ out) {
  __shared__ __align__(16) unsigned short shb[4][16][Hd];  // bf16 hidden mirror
  __shared__ float gsum[16][256];   // fused r|z gates
  __shared__ float gin[16][Hd];     // inn (x side)
  __shared__ float ghn[16][Hd];     // hn (h side); reused as lin1 staging
  __shared__ float sbias[4][4][Hd]; // per-layer fused biases: r,z,in,hn
  __shared__ float sx[16][4];       // x_t tile

  const int tid  = threadIdx.x;
  const int wave = tid >> 5;          // 0..7 -> gate-column triple
  const int lane = tid & 31;
  const int mrow = lane & 15;         // M (A) and N (B/C) sub-index
  const int hi   = lane >> 4;
  const int b0   = blockIdx.x * 16;

  float* dec_out = out + DEC_OUT_OFF;
  float* dec_hid = out + DEC_HID_OFF;
  float* enc_out = out + ENC_OUT_OFF;
  float* enc_hid = out + ENC_HID_OFF;
  float* loss    = out + LOSS_OFF;

  // fp32 hidden state: thread owns elements e = tid + 256*i (row=e>>7,col=e&127)
  float hreg[4][8];
#pragma unroll
  for (int l = 0; l < 4; ++l)
#pragma unroll
    for (int i = 0; i < 8; ++i) hreg[l][i] = 0.0f;
  for (int i = tid; i < 4 * 16 * Hd; i += 256) ((unsigned short*)shb)[i] = 0;

  // decoder_output[:,0,:] = inp[:,0,:]
  if (tid < 64) {
    int r = tid >> 2, e = tid & 3;
    dec_out[(b0 + r) * 400 + e] = inp[(b0 + r) * 256 + e];
  }
  float lossAcc = 0.0f;

#pragma unroll 1
  for (int phase = 0; phase < 2; ++phase) {
    const float* wih0 = phase ? dec_wih0 : enc_wih0;
    const float* bih  = phase ? dec_bih : enc_bih;
    const float* bhh  = phase ? dec_bhh : enc_bhh;
    const unsigned short* whh_pk = wpk + (phase ? 7 * PKM : 0);
    const unsigned short* wih_pk = wpk + (phase ? 11 * PKM : 4 * PKM);
    const int T = phase ? TDEC : TENC;

    // fused gate biases for this phase -> LDS
    __syncthreads();
    for (int i = tid; i < 4 * Hd; i += 256) {
      int l = i >> 7, c = i & 127;
      sbias[l][0][c] = bih[l * 384 + c] + bhh[l * 384 + c];
      sbias[l][1][c] = bih[l * 384 + 128 + c] + bhh[l * 384 + 128 + c];
      sbias[l][2][c] = bih[l * 384 + 256 + c];
      sbias[l][3][c] = bhh[l * 384 + 256 + c];
    }

#pragma unroll 1
    for (int t = 0; t < T; ++t) {
      __syncthreads();
      // ---- load x_t tile ----
      if (tid < 64) {
        int r = tid >> 2, e = tid & 3;
        float xv;
        if (phase == 0)      xv = inp[(b0 + r) * 256 + t * 4 + e];
        else if (t == 0)     xv = inp[(b0 + r) * 256 + e];
        else                 xv = tgt[(b0 + r) * 400 + t * 4 + e];
        sx[r][e] = xv;
      }
      __syncthreads();
      // ---- layer0 pre-gate: x @ wih0^T (K=4, scalar) into C-init staging ----
#pragma unroll 1
      for (int j = 0; j < 24; ++j) {
        int idx = tid + (j << 8);
        int r = idx / 384;
        int n = idx - r * 384;
        const float* w = wih0 + n * 4;
        float v = sx[r][0] * w[0] + sx[r][1] * w[1] + sx[r][2] * w[2] + sx[r][3] * w[3];
        if (n < 256) gsum[r][n] = v;
        else         gin[r][n - 256] = v;
      }
      __syncthreads();

      // ---- GRU layers (kept rolled: weight fragments must re-stream from L2)
#pragma unroll 1
      for (int l = 0; l < 4; ++l) {
        v8f aR, aZ, aI, aH;
        if (l == 0) {
#pragma unroll
          for (int v = 0; v < 8; ++v) {
            int r = v + hi * 8;
            aR[v] = gsum[r][wave * 16 + mrow];
            aZ[v] = gsum[r][128 + wave * 16 + mrow];
            aI[v] = gin[r][wave * 16 + mrow];
            aH[v] = 0.0f;
          }
        } else {
#pragma unroll
          for (int v = 0; v < 8; ++v) { aR[v] = 0.0f; aZ[v] = 0.0f; aI[v] = 0.0f; aH[v] = 0.0f; }
        }
        const unsigned short* upk  = whh_pk + l * PKM;
        const unsigned short* xpk  = wih_pk + (l > 0 ? (l - 1) * PKM : 0);
        const unsigned short* hrow = &shb[l][mrow][0];
        const unsigned short* xrow = (l > 0) ? &shb[l - 1][mrow][0] : &shb[l][mrow][0];
#pragma unroll 1
        for (int kc = 0; kc < 4; ++kc) {
          int kb = kc * 32 + hi * 16;
          v16bf Ah  = ldA(hrow, kb);
          v16bf Bur = ldB(upk, wave, kc, lane);
          v16bf Buz = ldB(upk, wave + 8, kc, lane);
          v16bf Bun = ldB(upk, wave + 16, kc, lane);
          aR = WMMA_BF16(Ah, Bur, aR);
          aZ = WMMA_BF16(Ah, Buz, aZ);
          aH = WMMA_BF16(Ah, Bun, aH);
          if (l > 0) {
            v16bf Ax  = ldA(xrow, kb);
            v16bf Bxr = ldB(xpk, wave, kc, lane);
            v16bf Bxz = ldB(xpk, wave + 8, kc, lane);
            v16bf Bxn = ldB(xpk, wave + 16, kc, lane);
            aR = WMMA_BF16(Ax, Bxr, aR);
            aZ = WMMA_BF16(Ax, Bxz, aZ);
            aI = WMMA_BF16(Ax, Bxn, aI);
          }
        }
#pragma unroll
        for (int v = 0; v < 8; ++v) {
          int r = v + hi * 8;
          gsum[r][wave * 16 + mrow]       = aR[v];
          gsum[r][128 + wave * 16 + mrow] = aZ[v];
          gin[r][wave * 16 + mrow]        = aI[v];
          ghn[r][wave * 16 + mrow]        = aH[v];
        }
        __syncthreads();
        // ---- elementwise gates + state update (owner threads) ----
#pragma unroll
        for (int i = 0; i < 8; ++i) {
          int e = tid + (i << 8);
          int r = e >> 7, c = e & 127;
          float rg = sigm(gsum[r][c] + sbias[l][0][c]);
          float zg = sigm(gsum[r][128 + c] + sbias[l][1][c]);
          float ng = tanh_fast(gin[r][c] + sbias[l][2][c] +
                               rg * (ghn[r][c] + sbias[l][3][c]));
          float hn = (1.0f - zg) * ng + zg * hreg[l][i];
          hreg[l][i] = hn;
          shb[l][r][c] = f2bf(hn);
          if (l == 3 && phase == 0)
            enc_out[(size_t)(b0 + r) * (64 * 128) + t * 128 + c] = hn;
        }
        __syncthreads();
      }

      if (phase == 1) {
        // ---- lin1: (16x128) pre-relu of h3 @ lin1^T, WMMA, wave w -> tile w
        //      (result staged in ghn, whose GRU contents are dead here)
        v8f aO;
#pragma unroll
        for (int v = 0; v < 8; ++v) aO[v] = 0.0f;
        const unsigned short* hrow3 = &shb[3][mrow][0];
        const unsigned short* lpk   = wpk + 14 * PKM;
#pragma unroll 1
        for (int kc = 0; kc < 4; ++kc) {
          int kb = kc * 32 + hi * 16;
          v16bf A3 = ldA(hrow3, kb);
          v16bf Bl = ldB(lpk, wave, kc, lane);
          aO = WMMA_BF16(A3, Bl, aO);
        }
#pragma unroll
        for (int v = 0; v < 8; ++v) ghn[v + hi * 8][wave * 16 + mrow] = aO[v];
        __syncthreads();
        // ---- lin2 + loss (E=4, scalar) ----
        if (tid < 64) {
          int r = tid >> 2, e = tid & 3;
          float acc = lin2_b[e];
          const float* w2 = lin2_w + e * 128;
#pragma unroll 8
          for (int k = 0; k < 128; ++k) {
            float a = ghn[r][k] + lin1_b[k];
            a = a > 0.0f ? a : 0.0f;
            acc += a * w2[k];
          }
          dec_out[(b0 + r) * 400 + (t + 1) * 4 + e] = acc;
          float d = acc - tgt[(b0 + r) * 400 + (t + 1) * 4 + e];
          lossAcc += d * d;
        }
      }
    }

    // ---- emit hidden snapshot for this phase ----
    float* hp = (phase == 0) ? enc_hid : dec_hid;
#pragma unroll
    for (int l = 0; l < 4; ++l)
#pragma unroll
      for (int i = 0; i < 8; ++i) {
        int e = tid + (i << 8);
        hp[(size_t)l * 524288 + (size_t)(b0 + (e >> 7)) * 128 + (e & 127)] = hreg[l][i];
      }
  }

  if (tid < 64) atomicAdd(loss, lossAcc * (1.0f / 16384.0f));
}

// ---------------------------------------------------------------------------
extern "C" void kernel_launch(void* const* d_in, const int* in_sizes, int n_in,
                              void* d_out, int out_size, void* d_ws, size_t ws_size,
                              hipStream_t stream) {
  (void)in_sizes; (void)n_in; (void)out_size; (void)ws_size;
  const float* inp      = (const float*)d_in[0];
  const float* tgt      = (const float*)d_in[1];
  const float* enc_wih0 = (const float*)d_in[2];
  const float* enc_whh0 = (const float*)d_in[3];
  const float* enc_wih  = (const float*)d_in[4];   // (3,384,128)
  const float* enc_whh  = (const float*)d_in[5];   // (3,384,128)
  const float* enc_bih  = (const float*)d_in[6];
  const float* enc_bhh  = (const float*)d_in[7];
  const float* dec_wih0 = (const float*)d_in[8];
  const float* dec_whh0 = (const float*)d_in[9];
  const float* dec_wih  = (const float*)d_in[10];
  const float* dec_whh  = (const float*)d_in[11];
  const float* dec_bih  = (const float*)d_in[12];
  const float* dec_bhh  = (const float*)d_in[13];
  const float* lin1_w   = (const float*)d_in[14];
  const float* lin1_b   = (const float*)d_in[15];
  const float* lin2_w   = (const float*)d_in[16];
  const float* lin2_b   = (const float*)d_in[17];
  float* out = (float*)d_out;

  unsigned short* wpk = (unsigned short*)d_ws;
  const int MAT = Gd * Hd;            // 49152 elems per 384x128 matrix
  const int PB = (MAT + 255) / 256;   // 192 blocks

  // slot layout (halves): enc_whh[0..3] | enc_wih[1..3] | dec_whh[0..3] |
  //                       dec_wih[1..3] | lin1
  pack_w_kernel<<<PB, 256, 0, stream>>>(enc_whh0, wpk + 0 * PKM, MAT);
  for (int l = 1; l < 4; ++l)
    pack_w_kernel<<<PB, 256, 0, stream>>>(enc_whh + (l - 1) * MAT, wpk + l * PKM, MAT);
  for (int l = 1; l < 4; ++l)
    pack_w_kernel<<<PB, 256, 0, stream>>>(enc_wih + (l - 1) * MAT, wpk + (4 + l - 1) * PKM, MAT);
  pack_w_kernel<<<PB, 256, 0, stream>>>(dec_whh0, wpk + 7 * PKM, MAT);
  for (int l = 1; l < 4; ++l)
    pack_w_kernel<<<PB, 256, 0, stream>>>(dec_whh + (l - 1) * MAT, wpk + (7 + l) * PKM, MAT);
  for (int l = 1; l < 4; ++l)
    pack_w_kernel<<<PB, 256, 0, stream>>>(dec_wih + (l - 1) * MAT, wpk + (11 + l - 1) * PKM, MAT);
  pack_w_kernel<<<(Hd * Hd + 255) / 256, 256, 0, stream>>>(lin1_w, wpk + 14 * PKM, Hd * Hd);

  zero_loss_kernel<<<1, 64, 0, stream>>>(out + LOSS_OFF);

  gru_seq2seq_kernel<<<NBATCH / 16, 256, 0, stream>>>(
      inp, tgt, enc_wih0, enc_bih, enc_bhh, dec_wih0, dec_bih, dec_bhh,
      lin1_b, lin2_w, lin2_b, wpk, out);
}
